// CRFlayer_21088289423548
// MI455X (gfx1250) — matrix operand order repaired and checked by reading, not compiled, not run on hardware
//
#include <hip/hip_runtime.h>
#include <hip/hip_bf16.h>
#include <stdint.h>

typedef __attribute__((ext_vector_type(16))) _Float16 v16h;
typedef __attribute__((ext_vector_type(8)))  float    v8f;

#define B_DIM 128
#define S_DIM 512
#define L_DIM 64
#define M_TOTAL (B_DIM * S_DIM)        // 65536 rows of (b,s)
#define TILES_M (M_TOTAL / 16)         // 4096 16-row WMMA tiles

// ---------------------------------------------------------------------------
// ws layout: floats [0..63] alpha partials, [64] score accumulator,
//            byte offset 512: 4096 f16 = exp(T)^T pre-packed in WMMA B-operand
//            lane order (8 groups of (nt,kh) x 32 lanes x 16 halfs).
// ---------------------------------------------------------------------------

// One 256-thread block: zero accumulators + build the B-operand image of
// exp(T)^T so the GEMM waves do zero setup transcendentals.
__global__ __launch_bounds__(256) void crf_prep(
    const float* __restrict__ transitions, float* __restrict__ ws)
{
    if (threadIdx.x < 65) ws[threadIdx.x] = 0.0f;

    _Float16* E16 = (_Float16*)(ws + 128);
    const int lane = threadIdx.x & 31;
    const int g    = threadIdx.x >> 5;       // g = nt*2 + kh  (8 groups)
    const int nt   = g >> 1;
    const int kh   = g & 1;
    // B[k][n] = exp(T[n][k]); lane holds col n = lane&15,
    // element e -> k = kh*32 + (lane>>4)*8 + (e&7) + ((e>>3)<<4)  (ISA 16-bit B striping)
    const int n     = nt * 16 + (lane & 15);
    const int kbase = kh * 32 + (lane >> 4) * 8;
    const float* src = transitions + n * L_DIM + kbase;
    #pragma unroll
    for (int e = 0; e < 16; ++e)
        E16[(g * 32 + lane) * 16 + e] = (_Float16)__expf(src[(e & 7) + ((e >> 3) << 4)]);
}

// ---------------------------------------------------------------------------
// alpha[i] = emit[0,0,i] + sum_{s>=1, mask} log( (exp(Emit) . exp(T)^T)[m,i] )
// Per wave: resident B operand (v16h x8 from ws), stream 16-row A tiles,
// 8x v_wmma_f32_16x16x32_f16 per tile, fused log4-grouped masked reduction.
// ---------------------------------------------------------------------------
__global__ __launch_bounds__(256) void crf_alpha_wmma(
    const float* __restrict__ emit,          // (M_TOTAL, 64)
    const uint8_t* __restrict__ mask,        // (B,S) bool, flat = m index
    const float* __restrict__ ws,            // E16 image at ws+128 floats
    float* __restrict__ ws_alpha)            // 64 floats (atomic accum)
{
    const int lane = threadIdx.x & 31;
    const int wave = threadIdx.x >> 5;
    const int hi   = lane >> 4;
    const int lo   = lane & 15;

    // ---- B operand: straight 32B-aligned vector loads, no VALU setup ----
    const v16h* Ep = (const v16h*)(ws + 128);
    v16h Bt[4][2];
    #pragma unroll
    for (int nt = 0; nt < 4; ++nt)
        #pragma unroll
        for (int kh = 0; kh < 2; ++kh)
            Bt[nt][kh] = Ep[(nt * 2 + kh) * 32 + lane];

    float accsum[4] = {0.f, 0.f, 0.f, 0.f};

    const int wslots = gridDim.x * 8;
    for (int t = blockIdx.x * 8 + wave; t < TILES_M; t += wslots) {
        const int mbase = t * 16;

        // 8 row flags for this lane's C/D rows (v + 8*hi): one b64 mask load
        const uint64_t mrow = *(const uint64_t*)(mask + mbase + 8 * hi);
        const bool seq0 = ((mbase & (S_DIM - 1)) == 0) && (hi == 0);
        bool f[8];
        #pragma unroll
        for (int v = 0; v < 8; ++v)
            f[v] = (((mrow >> (8 * v)) & 0xffull) != 0) && !(seq0 && v == 0);

        // ---- A operand: 8x b128 loads; per-lane K runs are 16B aligned.
        // ISA 16-bit A mapping: e -> k = (e&7) + ((e>>3)<<4) + (hi<<3) (+32)
        // => runs of 8 floats at k-offsets {0,16,32,48} + hi*8.
        const float* ebase = emit + (size_t)(mbase + lo) * L_DIM + hi * 8;
        float av[32];
        #pragma unroll
        for (int r = 0; r < 4; ++r) {
            const float4 p0 = *(const float4*)(ebase + r * 16);
            const float4 p1 = *(const float4*)(ebase + r * 16 + 4);
            av[r * 8 + 0] = p0.x; av[r * 8 + 1] = p0.y;
            av[r * 8 + 2] = p0.z; av[r * 8 + 3] = p0.w;
            av[r * 8 + 4] = p1.x; av[r * 8 + 5] = p1.y;
            av[r * 8 + 6] = p1.z; av[r * 8 + 7] = p1.w;
        }

        // Prefetch next tile's rows (global_prefetch_b8)
        const int tn = t + wslots;
        if (tn < TILES_M)
            __builtin_prefetch(emit + (size_t)(tn * 16 + lo) * L_DIM + hi * 8, 0, 0);

        v16h At[2];
        #pragma unroll
        for (int kh = 0; kh < 2; ++kh) {
            v16h tmp;
            #pragma unroll
            for (int e = 0; e < 16; ++e)
                tmp[e] = (_Float16)__expf(av[(kh * 2 + (e >> 3)) * 8 + (e & 7)]);
            At[kh] = tmp;
        }

        // ---- 4 N-tiles x (K=64 = 2x K32 WMMA), log grouped 4-wide ----
        #pragma unroll
        for (int nt = 0; nt < 4; ++nt) {
            v8f acc = {};
            acc = __builtin_amdgcn_wmma_f32_16x16x32_f16(
                false, At[0], false, Bt[nt][0], (short)0, acc, false, false);
            acc = __builtin_amdgcn_wmma_f32_16x16x32_f16(
                false, At[1], false, Bt[nt][1], (short)0, acc, false, false);
            // log(prod of 4) == sum of logs; masked rows contribute 1.0.
            float p0 = (f[0] ? acc[0] : 1.0f) * (f[1] ? acc[1] : 1.0f)
                     * (f[2] ? acc[2] : 1.0f) * (f[3] ? acc[3] : 1.0f);
            float p1 = (f[4] ? acc[4] : 1.0f) * (f[5] ? acc[5] : 1.0f)
                     * (f[6] ? acc[6] : 1.0f) * (f[7] ? acc[7] : 1.0f);
            accsum[nt] += __logf(p0) + __logf(p1);
        }
    }

    #pragma unroll
    for (int nt = 0; nt < 4; ++nt)
        atomicAdd(&ws_alpha[nt * 16 + lo], accsum[nt]);
}

// ---------------------------------------------------------------------------
// Gold-path score per batch + start/end transitions.
// ---------------------------------------------------------------------------
__global__ __launch_bounds__(256) void crf_score(
    const float* __restrict__ emit,
    const int* __restrict__ labels,
    const uint8_t* __restrict__ mask,
    const float* __restrict__ transitions,
    const float* __restrict__ strans,
    const float* __restrict__ etrans,
    float* __restrict__ ws_score)
{
    __shared__ float ssum[256];
    __shared__ int   scnt[256];
    const int b   = blockIdx.x;
    const int tid = threadIdx.x;

    float psum = 0.f;
    int   pcnt = 0;
    for (int s = tid; s < S_DIM; s += 256) {
        if (mask[b * S_DIM + s]) {
            const int lab = labels[b * S_DIM + s];
            float v = emit[((size_t)b * S_DIM + s) * L_DIM + lab];
            if (s > 0) {
                const int lp = labels[b * S_DIM + s - 1];
                v += transitions[lp * L_DIM + lab];
            }
            psum += v;
            ++pcnt;
        }
    }
    ssum[tid] = psum;
    scnt[tid] = pcnt;
    __syncthreads();
    for (int off = 128; off > 0; off >>= 1) {
        if (tid < off) { ssum[tid] += ssum[tid + off]; scnt[tid] += scnt[tid + off]; }
        __syncthreads();
    }
    if (tid == 0) {
        const int ends = scnt[0] - 1;
        const float sc = ssum[0]
                       + strans[labels[b * S_DIM]]
                       + etrans[labels[b * S_DIM + ends]];
        atomicAdd(ws_score, sc);
    }
}

// ---------------------------------------------------------------------------
// out = (logsumexp(alpha) - score) / B ; alpha[i] = ws[i] + emit[0,0,i]
// ---------------------------------------------------------------------------
__global__ void crf_finalize(const float* __restrict__ emit,
                             const float* __restrict__ ws,
                             float* __restrict__ out)
{
    float a[L_DIM];
    float m = -1e30f;
    #pragma unroll
    for (int i = 0; i < L_DIM; ++i) {
        a[i] = ws[i] + emit[i];
        m = fmaxf(m, a[i]);
    }
    float sum = 0.f;
    #pragma unroll
    for (int i = 0; i < L_DIM; ++i)
        sum += __expf(a[i] - m);
    const float logZ = m + __logf(sum);
    out[0] = (logZ - ws[L_DIM]) / (float)B_DIM;
}

extern "C" void kernel_launch(void* const* d_in, const int* in_sizes, int n_in,
                              void* d_out, int out_size, void* d_ws, size_t ws_size,
                              hipStream_t stream) {
    const float*   emit        = (const float*)d_in[0];
    const int*     labels      = (const int*)d_in[1];
    const uint8_t* mask        = (const uint8_t*)d_in[2];
    const float*   transitions = (const float*)d_in[3];
    const float*   strans      = (const float*)d_in[4];
    const float*   etrans      = (const float*)d_in[5];
    float* ws  = (float*)d_ws;
    float* out = (float*)d_out;

    crf_prep<<<1, 256, 0, stream>>>(transitions, ws);
    // 256 blocks x 8 waves = 2048 wave slots; 4096 tiles -> 2 tiles/wave.
    crf_alpha_wmma<<<256, 256, 0, stream>>>(emit, mask, ws, ws);
    crf_score<<<B_DIM, 256, 0, stream>>>(emit, labels, mask, transitions,
                                         strans, etrans, ws + L_DIM);
    crf_finalize<<<1, 1, 0, stream>>>(emit, ws, out);
}
